// Attention_Conv_surface_41669772706181
// MI455X (gfx1250) — matrix operand order repaired and checked by reading, not compiled, not run on hardware
//
#include <hip/hip_runtime.h>
#include <hip/hip_bf16.h>

typedef __attribute__((ext_vector_type(16))) _Float16 v16h;
typedef __attribute__((ext_vector_type(8)))  float    v8f;
typedef unsigned int u32x4 __attribute__((ext_vector_type(4)));
typedef int          i32x8 __attribute__((ext_vector_type(8)));
typedef int          i32x4 __attribute__((ext_vector_type(4)));

union AF {            // one WMMA f16 operand fragment: 16 halves / lane = 8 VGPRs
    v16h v;
    _Float16 e[16];
    uint4 q[2];
};

#define WMMA_F16(A, B, C) \
    __builtin_amdgcn_wmma_f32_16x16x32_f16(false, (A), false, (B), (short)0, (C), false, false)

#if defined(__has_builtin)
#if __has_builtin(__builtin_amdgcn_tensor_load_to_lds) && __has_builtin(__builtin_amdgcn_s_wait_tensorcnt)
#define USE_TDM 1
#endif
#endif

constexpr int BSZ = 4, NV = 2048, NN = 32, NK = 64, NH = 4, DK = 16;
constexpr int ROWS = BSZ * NV;               // 8192
constexpr int KB   = 128;                    // keys staged per LDS block
constexpr float L2E = 1.44269504088896340736f;

// ---------------------------------------------------------------- weights->f16
__global__ void wconv_kernel(const float* __restrict__ wq, const float* __restrict__ wk,
                             const float* __restrict__ wv, const float* __restrict__ wo,
                             _Float16* __restrict__ wh) {
    const float* ws[4] = {wq, wk, wv, wo};
    for (int m = 0; m < 4; ++m)
        for (int i = threadIdx.x; i < NK * NK; i += blockDim.x)
            wh[m * NK * NK + i] = (_Float16)ws[m][i];
}

// ---------------------------------------------------------------- theta (GCN feature)
// out[b,v,k] = sum_s max_n relu(dirn[b,v,n] . normalize(dirs[:, s*K+k]))
__global__ void theta_kernel(const int* __restrict__ nbr, const float* __restrict__ verts,
                             const float* __restrict__ qd, const float* __restrict__ kd,
                             const float* __restrict__ vd, _Float16* __restrict__ th) {
    __shared__ float dirn[NN][3];
    __shared__ float cm[256];
    const int t  = threadIdx.x;
    const int bv = blockIdx.x;                 // b*V + v
    const int b  = bv >> 11;

    if (t < NN) {
        int idx = nbr[bv * NN + t];
        const float* pn = verts + ((size_t)b * NV + idx) * 3;
        const float* pc = verts + (size_t)bv * 3;
        float dx = pn[0] - pc[0], dy = pn[1] - pc[1], dz = pn[2] - pc[2];
        float n  = sqrtf(dx * dx + dy * dy + dz * dz);
        float inv = 1.0f / fmaxf(n, 1e-12f);
        dirn[t][0] = dx * inv; dirn[t][1] = dy * inv; dirn[t][2] = dz * inv;
    }
    __syncthreads();

    const float* dsets[3] = {qd, kd, vd};
    for (int set = 0; set < 3; ++set) {
        const float* d = dsets[set];
        float a = d[t], bb = d[256 + t], c = d[512 + t];
        float len = sqrtf(a * a + bb * bb + c * c);
        float inv = 1.0f / fmaxf(len, 1e-12f);
        a *= inv; bb *= inv; c *= inv;
        float m = 0.0f;                         // relu folded into max (m >= 0)
        #pragma unroll
        for (int n = 0; n < NN; ++n)
            m = fmaxf(m, dirn[n][0] * a + dirn[n][1] * bb + dirn[n][2] * c);
        if (set) __syncthreads();
        cm[t] = m;
        __syncthreads();
        if (t < NK) {
            float s = cm[t] + cm[NK + t] + cm[2 * NK + t] + cm[3 * NK + t];
            th[((size_t)set * ROWS + bv) * NK + t] = (_Float16)s;
        }
    }
}

// ---------------------------------------------------------------- q/k/v projection (WMMA)
__global__ void proj_qkv_kernel(const _Float16* __restrict__ th, const _Float16* __restrict__ wh,
                                const float* __restrict__ bq, const float* __restrict__ bk,
                                const float* __restrict__ bv,
                                _Float16* __restrict__ qh, _Float16* __restrict__ kh,
                                _Float16* __restrict__ vt) {
    const int set  = blockIdx.y;
    const int lane = threadIdx.x & 31;
    const int wave = threadIdx.x >> 5;
    const int tile = blockIdx.x * 8 + wave;
    const int rowbase = tile * 16;
    const bool lo  = lane < 16;
    const int lh   = lane & 15;

    const _Float16* X = th + (size_t)set * ROWS * NK;
    const _Float16* W = wh + set * NK * NK;
    const float* bias = (set == 0) ? bq : ((set == 1) ? bk : bv);

    const int arow = rowbase + lh;
    const int koff = lo ? 0 : 8;
    const uint4* Xq = reinterpret_cast<const uint4*>(X + (size_t)arow * NK + koff);
    AF a0, a1;
    a0.q[0] = Xq[0]; a0.q[1] = Xq[2];
    a1.q[0] = Xq[4]; a1.q[1] = Xq[6];

    AF bf[4][2];
    #pragma unroll
    for (int nt = 0; nt < 4; ++nt)
        #pragma unroll
        for (int st = 0; st < 2; ++st) {
            const uint4* Wq4 = reinterpret_cast<const uint4*>(
                W + (size_t)(nt * 16 + lh) * NK + st * 32 + (lo ? 0 : 8));
            bf[nt][st].q[0] = Wq4[0]; bf[nt][st].q[1] = Wq4[2];
        }

    v8f acc[4] = {};
    #pragma unroll
    for (int nt = 0; nt < 4; ++nt) {
        acc[nt] = WMMA_F16(a0.v, bf[nt][0].v, acc[nt]);
        acc[nt] = WMMA_F16(a1.v, bf[nt][1].v, acc[nt]);
    }

    #pragma unroll
    for (int nt = 0; nt < 4; ++nt) {
        const int n = nt * 16 + lh;            // h = nt, d = lh
        const float bval = bias[n];
        #pragma unroll
        for (int i = 0; i < 8; ++i) {
            const int r  = rowbase + i + (lo ? 0 : 8);
            const int bg = r >> 11;
            const int vv = r & (NV - 1);
            const float val = acc[nt][i] + bval;
            if (set == 0)
                qh[(((size_t)bg * NH + nt) * NV + vv) * DK + lh] = (_Float16)(val * 0.25f);
            else if (set == 1)
                kh[(((size_t)bg * NH + nt) * NV + vv) * DK + lh] = (_Float16)val;
            else
                vt[(((size_t)bg * NH + nt) * DK + lh) * NV + vv] = (_Float16)val;
        }
    }
}

// ---------------------------------------------------------------- TDM 2D tile load (global -> LDS)
#ifdef USE_TDM
__device__ __forceinline__ void tdm_load_2d(uint32_t lds_addr, const void* gptr,
                                            uint32_t tile_d0_dw, uint32_t tile_d1,
                                            uint32_t tensor_d0_dw, uint32_t tensor_d1,
                                            uint32_t stride_dw) {
    const uint64_t ga = (uint64_t)(uintptr_t)gptr;
    u32x4 g0;
    g0[0] = 1u;                                              // count=1, user descriptor
    g0[1] = lds_addr;                                        // LDS byte address
    g0[2] = (uint32_t)ga;                                    // global_addr[31:0]
    g0[3] = (uint32_t)((ga >> 32) & 0x01FFFFFFu) | (2u << 30); // global_addr[56:32] | type=2
    i32x8 g1;
    g1[0] = (int)(2u << 16);                                 // data_size=2 (4-byte elements)
    g1[1] = (int)((tensor_d0_dw & 0xFFFFu) << 16);           // tensor_dim0[15:0]
    g1[2] = (int)(((tensor_d0_dw >> 16) & 0xFFFFu) | ((tensor_d1 & 0xFFFFu) << 16));
    g1[3] = (int)(((tensor_d1 >> 16) & 0xFFFFu) | ((tile_d0_dw & 0xFFFFu) << 16));
    g1[4] = (int)(tile_d1 & 0xFFFFu);                        // tile_dim1 (tile_dim2=0)
    g1[5] = (int)stride_dw;                                  // tensor_dim0_stride[31:0]
    g1[6] = 0;
    g1[7] = 0;
    const i32x4 z4 = {0, 0, 0, 0};                           // D# groups 2/3 unused (<=2D)
    const i32x8 z8 = {0, 0, 0, 0, 0, 0, 0, 0};
    __builtin_amdgcn_tensor_load_to_lds(g0, g1, z4, z4, z8, 0);
}
#endif

// ---------------------------------------------------------------- flash attention
// 4 waves/block, 16 query rows/wave; K/V staged per 128-key block into LDS,
// double-buffered via the Tensor Data Mover (TENSORcnt).
__global__ void attn_kernel(const _Float16* __restrict__ qh, const _Float16* __restrict__ kh,
                            const _Float16* __restrict__ vt, _Float16* __restrict__ xh) {
    __shared__ alignas(16) _Float16 Kbuf[2][KB * DK];        // 128 rows x 16 halves
    __shared__ alignas(16) _Float16 Vbuf[2][DK * KB];        // 16 rows x 128 halves
    __shared__ alignas(16) _Float16 plds[4][16][32];

    const int lane = threadIdx.x & 31;
    const int wave = threadIdx.x >> 5;
    const bool lo  = lane < 16;
    const int lh   = lane & 15;

    const int blk = blockIdx.x;
    const int bh  = blk >> 5;                   // 32 row-blocks per (b,h)
    const int rt  = blk & 31;
    const int b   = bh >> 2;
    const int h   = bh & 3;
    const int qbase = rt * 64 + wave * 16;

    const _Float16* Qp = qh + (size_t)bh * NV * DK;
    const _Float16* Kp = kh + (size_t)bh * NV * DK;
    const _Float16* Vp = vt + (size_t)bh * DK * NV;

    AF aQ;
    {
        const uint4* p = reinterpret_cast<const uint4*>(Qp + (size_t)(qbase + lh) * DK);
        aQ.q[0] = lo ? p[0] : make_uint4(0u, 0u, 0u, 0u);
        aQ.q[1] = lo ? p[1] : make_uint4(0u, 0u, 0u, 0u);
    }

    float m[8], l[8];
    #pragma unroll
    for (int i = 0; i < 8; ++i) { m[i] = -1e30f; l[i] = 0.0f; }
    v8f acc = {};

    _Float16* P = &plds[wave][0][0];
    const int nkb = NV / KB;                    // 16 key blocks

    // ---- stage helper: issue K+V DMA for key block kb into buffer p ----
    auto stage = [&](int kb, int p) {
#ifdef USE_TDM
        if (wave == 0) {
            tdm_load_2d((uint32_t)(uintptr_t)&Kbuf[p][0], Kp + (size_t)kb * KB * DK,
                        /*tile_d0=*/DK / 2, /*tile_d1=*/KB,
                        /*tensor_d0=*/DK / 2, /*tensor_d1=*/NV, /*stride=*/DK / 2);
            tdm_load_2d((uint32_t)(uintptr_t)&Vbuf[p][0], Vp + (size_t)kb * KB,
                        /*tile_d0=*/KB / 2, /*tile_d1=*/DK,
                        /*tensor_d0=*/NV / 2, /*tensor_d1=*/DK, /*stride=*/NV / 2);
        }
#else
        // cooperative fallback: 128 threads copy 4KB + 4KB
        const uint4* gk = reinterpret_cast<const uint4*>(Kp + (size_t)kb * KB * DK);
        uint4* lk = reinterpret_cast<uint4*>(&Kbuf[p][0]);
        for (int i = threadIdx.x; i < KB * DK / 8; i += 128) lk[i] = gk[i];
        uint4* lv = reinterpret_cast<uint4*>(&Vbuf[p][0]);
        const _Float16* gv = Vp + (size_t)kb * KB;
        for (int i = threadIdx.x; i < DK * KB / 8; i += 128) {
            int row = i >> 4, col = i & 15;     // 16 uint4 per 128-half row
            lv[i] = reinterpret_cast<const uint4*>(gv + (size_t)row * NV)[col];
        }
#endif
    };

    stage(0, 0);                                // prologue

    for (int kb = 0; kb < nkb; ++kb) {
        const int p = kb & 1;
#ifdef USE_TDM
        if (wave == 0) __builtin_amdgcn_s_wait_tensorcnt(0);
#endif
        __syncthreads();                        // staged data visible; prev block consumed
        if (kb + 1 < nkb) stage(kb + 1, p ^ 1); // overlap DMA of next block with compute

        const _Float16* Kt = &Kbuf[p][0];
        const _Float16* Vt = &Vbuf[p][0];

        #pragma unroll
        for (int ks = 0; ks < KB; ks += 32) {
            // K^T B-fragments from LDS (keys ks..ks+15, ks+16..ks+31)
            AF bk0, bk1;
            if (lo) {
                const uint4* p0 = reinterpret_cast<const uint4*>(Kt + (ks + lh) * DK);
                const uint4* p1 = reinterpret_cast<const uint4*>(Kt + (ks + 16 + lh) * DK);
                bk0.q[0] = p0[0]; bk0.q[1] = p0[1];
                bk1.q[0] = p1[0]; bk1.q[1] = p1[1];
            } else {
                bk0.q[0] = bk0.q[1] = make_uint4(0u, 0u, 0u, 0u);
                bk1.q[0] = bk1.q[1] = make_uint4(0u, 0u, 0u, 0u);
            }
            // V B-fragment from LDS (transposed layout: contiguous keys per lane)
            AF bV;
            {
                const uint4* pv = reinterpret_cast<const uint4*>(Vt + lh * KB + ks + (lo ? 0 : 8));
                bV.q[0] = pv[0]; bV.q[1] = pv[2];
            }

            v8f c0 = {}, c1 = {};
            c0 = WMMA_F16(aQ.v, bk0.v, c0);
            c1 = WMMA_F16(aQ.v, bk1.v, c1);

            // online softmax: row i lives in VGPR i, cols striped over 16-lane halves
            #pragma unroll
            for (int i = 0; i < 8; ++i) {
                float s0 = c0[i], s1 = c1[i];
                float sm = fmaxf(s0, s1);
                #pragma unroll
                for (int off = 1; off < 16; off <<= 1) sm = fmaxf(sm, __shfl_xor(sm, off));
                const float mn = fmaxf(m[i], sm);
                const float sc = exp2f((m[i] - mn) * L2E);
                const float p0 = exp2f((s0 - mn) * L2E);
                const float p1 = exp2f((s1 - mn) * L2E);
                float rs = p0 + p1;
                #pragma unroll
                for (int off = 1; off < 16; off <<= 1) rs += __shfl_xor(rs, off);
                l[i] = l[i] * sc + rs;
                acc[i] *= sc;
                m[i] = mn;
                c0[i] = p0; c1[i] = p1;
            }

            // C-layout -> A-layout transpose of P through per-wave LDS tile
            #pragma unroll
            for (int i = 0; i < 8; ++i) {
                const int r = i + (lo ? 0 : 8);
                P[r * 32 + lh]      = (_Float16)c0[i];
                P[r * 32 + 16 + lh] = (_Float16)c1[i];
            }
            asm volatile("s_wait_dscnt 0" ::: "memory");
            AF aP;
            {
                const uint4* pp = reinterpret_cast<const uint4*>(P + lh * 32 + (lo ? 0 : 8));
                aP.q[0] = pp[0]; aP.q[1] = pp[2];
            }
            acc = WMMA_F16(aP.v, bV.v, acc);
        }
    }

    #pragma unroll
    for (int i = 0; i < 8; ++i) {
        const int r = qbase + i + (lo ? 0 : 8);
        const float val = acc[i] / l[i];
        xh[((size_t)b * NV + r) * NK + h * DK + lh] = (_Float16)val;
    }
}

// ---------------------------------------------------------------- output projection (WMMA, f32 out)
__global__ void proj_out_kernel(const _Float16* __restrict__ xh, const _Float16* __restrict__ wo,
                                const float* __restrict__ bo, float* __restrict__ out) {
    const int lane = threadIdx.x & 31;
    const int wave = threadIdx.x >> 5;
    const int tile = blockIdx.x * 8 + wave;
    const int rowbase = tile * 16;
    const bool lo = lane < 16;
    const int lh  = lane & 15;

    const int arow = rowbase + lh;
    const int koff = lo ? 0 : 8;
    const uint4* Xq = reinterpret_cast<const uint4*>(xh + (size_t)arow * NK + koff);
    AF a0, a1;
    a0.q[0] = Xq[0]; a0.q[1] = Xq[2];
    a1.q[0] = Xq[4]; a1.q[1] = Xq[6];

    AF bf[4][2];
    #pragma unroll
    for (int nt = 0; nt < 4; ++nt)
        #pragma unroll
        for (int st = 0; st < 2; ++st) {
            const uint4* Wq4 = reinterpret_cast<const uint4*>(
                wo + (size_t)(nt * 16 + lh) * NK + st * 32 + (lo ? 0 : 8));
            bf[nt][st].q[0] = Wq4[0]; bf[nt][st].q[1] = Wq4[2];
        }

    v8f acc[4] = {};
    #pragma unroll
    for (int nt = 0; nt < 4; ++nt) {
        acc[nt] = WMMA_F16(a0.v, bf[nt][0].v, acc[nt]);
        acc[nt] = WMMA_F16(a1.v, bf[nt][1].v, acc[nt]);
    }

    #pragma unroll
    for (int nt = 0; nt < 4; ++nt) {
        const int n = nt * 16 + lh;
        const float bval = bo[n];
        #pragma unroll
        for (int i = 0; i < 8; ++i) {
            const int r = rowbase + i + (lo ? 0 : 8);
            out[(size_t)r * NK + n] = acc[nt][i] + bval;
        }
    }
}

// ---------------------------------------------------------------- launch
extern "C" void kernel_launch(void* const* d_in, const int* in_sizes, int n_in,
                              void* d_out, int out_size, void* d_ws, size_t ws_size,
                              hipStream_t stream) {
    (void)in_sizes; (void)n_in; (void)out_size; (void)ws_size;
    const int*   nbr   = (const int*)  d_in[0];
    const float* verts = (const float*)d_in[1];
    const float* qd    = (const float*)d_in[2];
    const float* kd    = (const float*)d_in[3];
    const float* vd    = (const float*)d_in[4];
    const float* Wq    = (const float*)d_in[5];
    const float* bq    = (const float*)d_in[6];
    const float* Wk    = (const float*)d_in[7];
    const float* bk    = (const float*)d_in[8];
    const float* Wv    = (const float*)d_in[9];
    const float* bv    = (const float*)d_in[10];
    const float* Wo    = (const float*)d_in[11];
    const float* bo    = (const float*)d_in[12];
    float* out = (float*)d_out;

    _Float16* wh = (_Float16*)d_ws;                    // 4 * 64*64
    _Float16* th = wh + 4 * NK * NK;                   // 3 * 8192*64
    _Float16* qh = th + 3 * (size_t)ROWS * NK;
    _Float16* kh = qh + (size_t)ROWS * NK;
    _Float16* vt = kh + (size_t)ROWS * NK;             // transposed per head
    _Float16* xh = vt + (size_t)ROWS * NK;

    wconv_kernel<<<1, 256, 0, stream>>>(Wq, Wk, Wv, Wo, wh);
    theta_kernel<<<ROWS, 256, 0, stream>>>(nbr, verts, qd, kd, vd, th);
    dim3 gqkv(ROWS / (16 * 8), 3);
    proj_qkv_kernel<<<gqkv, 256, 0, stream>>>(th, wh, bq, bk, bv, qh, kh, vt);
    attn_kernel<<<BSZ * NH * (NV / 64), 128, 0, stream>>>(qh, kh, vt, xh);
    proj_out_kernel<<<ROWS / (16 * 8), 256, 0, stream>>>(xh, wh + 3 * NK * NK, bo, out);
}
